// SparseMultiHeadAttention_27255862460456
// MI455X (gfx1250) — matrix operand order repaired and checked by reading, not compiled, not run on hardware
//
#include <hip/hip_runtime.h>

// ---------------------------------------------------------------------------
// Types for CDNA5 WMMA (gfx1250, wave32)
// ---------------------------------------------------------------------------
typedef __attribute__((ext_vector_type(8)))  __bf16        bf8;
typedef __attribute__((ext_vector_type(16))) __bf16        v16bf;
typedef __attribute__((ext_vector_type(8)))  float         v8f;
typedef __attribute__((ext_vector_type(4)))  unsigned int  u32x4;
typedef __attribute__((ext_vector_type(8)))  unsigned short us8;
typedef unsigned short u16;
typedef unsigned int   u32;

#define D_MODEL 1024
#define NHEADS  16
#define DK      64
#define SEQ     2048
#define BATCH   4
#define KTOP    204           // int((1-0.9)*2048)
#define MTOT    (BATCH*SEQ)   // 8192

__device__ inline u16 f2bf(float f) {
    u32 u = __float_as_uint(f);
    u32 r = (u + 0x7FFFu + ((u >> 16) & 1u)) >> 16;   // RNE f32 -> bf16
    return (u16)r;
}
// order-preserving monotonic uint mapping of an IEEE f32
__device__ inline u32 ordf(float f) {
    u32 u = __float_as_uint(f);
    return u ^ ((u32)((int)u >> 31) | 0x80000000u);
}
__device__ inline v16bf cat16(bf8 lo, bf8 hi) {
    return __builtin_shufflevector(lo, hi, 0,1,2,3,4,5,6,7,8,9,10,11,12,13,14,15);
}
__device__ inline v8f wmma_bf16(v16bf a, v16bf b, v8f c) {
    return __builtin_amdgcn_wmma_f32_16x16x32_bf16(false, a, false, b, (short)0, c,
                                                   false, false);
}

// CDNA5 async global->LDS copy (ASYNCcnt-tracked DMA path).
// LDS operand: low 32 bits of the flat shared address are the LDS byte offset.
__device__ inline void async_load_b128(const void* gptr, void* lptr) {
    u32 lds = (u32)(__SIZE_TYPE__)lptr;
    unsigned long long ga = (unsigned long long)(__SIZE_TYPE__)gptr;
    asm volatile("global_load_async_to_lds_b128 %0, %1, off"
                 :: "v"(lds), "v"(ga) : "memory");
}
__device__ inline void async_wait0() {
    asm volatile("s_wait_asynccnt 0x0" ::: "memory");
}

// ---------------------------------------------------------------------------
// f32 -> bf16 conversion
// ---------------------------------------------------------------------------
__global__ void cvt_kernel(const float* __restrict__ src, u16* __restrict__ dst, int n) {
    int i = blockIdx.x * 256 + threadIdx.x;
    if (i < n) dst[i] = f2bf(src[i]);
}

// ---------------------------------------------------------------------------
// Tiled WMMA GEMM:  out[m,n] = sum_k A[m,k]*W[n,k] + bias[n]
// A: MTOT x 1024 bf16 row-major, W: 1024 x 1024 bf16 row-major (N x K)
// mode 0: write bf16 head-major [b][h][s][64]   (m=b*SEQ+s, n=h*64+d)
// mode 1: write f32 row-major   [m][1024]
// Block tile 128(M) x 64(N), BK=32, 256 threads = 8 waves (4x2), wave: 32x32.
// Double-buffered LDS staged via async global->LDS DMA (one barrier per step).
// ---------------------------------------------------------------------------
__global__ __launch_bounds__(256, 2)
void gemm_bf16(const u16* __restrict__ A, const u16* __restrict__ W,
               const float* __restrict__ bias, void* __restrict__ out, int mode) {
    __shared__ u16 At[2][128 * 32];
    __shared__ u16 Bt[2][64 * 32];

    const int tid  = threadIdx.x;
    const int lane = tid & 31;
    const int wid  = tid >> 5;           // 0..7
    const int m0   = blockIdx.y * 128;
    const int n0   = blockIdx.x * 64;
    const int wm   = wid >> 1;           // 0..3
    const int wn   = wid & 1;            // 0..1

    v8f acc[2][2];
    const v8f vzero = {0.f,0.f,0.f,0.f,0.f,0.f,0.f,0.f};
    acc[0][0] = vzero; acc[0][1] = vzero; acc[1][0] = vzero; acc[1][1] = vzero;

    const int rl = lane & 15;
    const int cq = lane >> 4;

    // staging assignment: A tile 128x32 -> thread: row ra, 16-half chunk ca (2x b128)
    //                     B tile  64x32 -> thread: row rb,  8-half chunk cb (1x b128)
    const int ra = tid >> 1, ca = (tid & 1) * 16;
    const int rb = tid >> 2, cb = (tid & 3) * 8;

    auto stage = [&](int kt, int buf) {
        const u16* ga = A + (size_t)(m0 + ra) * 1024 + kt + ca;
        const u16* gb = W + (size_t)(n0 + rb) * 1024 + kt + cb;
        async_load_b128(ga,     &At[buf][ra * 32 + ca]);
        async_load_b128(ga + 8, &At[buf][ra * 32 + ca + 8]);
        async_load_b128(gb,     &Bt[buf][rb * 32 + cb]);
    };

    stage(0, 0);
    for (int it = 0; it < 32; ++it) {
        async_wait0();          // our async writes landed in LDS
        __syncthreads();        // everyone's tile[it&1] visible
        if (it + 1 < 32) stage((it + 1) * 32, (it + 1) & 1);

        const u16* at = At[it & 1];
        const u16* bt = Bt[it & 1];
#pragma unroll
        for (int tm = 0; tm < 2; ++tm) {
            // A fragment 16x32: lane holds row rl, K chunks [8cq,+8) & [16+8cq,+8)
            const u16* ap = at + (wm * 32 + tm * 16 + rl) * 32;
            v16bf av = cat16(*(const bf8*)(ap + 8 * cq),
                             *(const bf8*)(ap + 16 + 8 * cq));
#pragma unroll
            for (int tn = 0; tn < 2; ++tn) {
                // B fragment 32x16: lane holds col rl, K chunks [8cq,+8) & [16+8cq,+8)
                const u16* bp = bt + (wn * 32 + tn * 16 + rl) * 32 + 8 * cq;
                v16bf bv = cat16(*(const bf8*)(bp), *(const bf8*)(bp + 16));
                acc[tm][tn] = wmma_bf16(av, bv, acc[tm][tn]);
            }
        }
    }

    // Epilogue: C layout: elem i of lane l -> M = i + 8*(l>>4), N = l&15
#pragma unroll
    for (int tm = 0; tm < 2; ++tm) {
#pragma unroll
        for (int tn = 0; tn < 2; ++tn) {
#pragma unroll
            for (int i = 0; i < 8; ++i) {
                int m = m0 + wm * 32 + tm * 16 + i + 8 * cq;
                int n = n0 + wn * 32 + tn * 16 + rl;
                float v = acc[tm][tn][i] + bias[n];
                if (mode == 0) {
                    int bb = m >> 11, s = m & 2047, hh = n >> 6, d = n & 63;
                    ((u16*)out)[(((size_t)(bb * NHEADS + hh)) * SEQ + s) * DK + d] = f2bf(v);
                } else {
                    ((float*)out)[(size_t)m * 1024 + n] = v;
                }
            }
        }
    }
}

// ---------------------------------------------------------------------------
// Sparse attention for one (b, h, 16-query tile).
// Qh/Kh/Vh: bf16 head-major [b][h][s][64].  ctx out: bf16 [b][s][1024].
// ---------------------------------------------------------------------------
__global__ __launch_bounds__(256)
void attn_kernel(const u16* __restrict__ Qh, const u16* __restrict__ Kh,
                 const u16* __restrict__ Vh, u16* __restrict__ ctx) {
    extern __shared__ char smem[];
    float* sc      = (float*)smem;                      // [16][2048] 131072 B
    float* partial = (float*)(smem + 131072);           // [8][16*64]  32768 B
    u16*   Vt      = (u16*)  (smem + 163840);           // [8][64*32]  32768 B
    u16*   Qt      = (u16*)  (smem + 196608);           // [16*64]      2048 B
    float* invden  = (float*)(smem + 198656);           // [16]           64 B
    u32*   hist    = (u32*)  (smem + 198720);           // [16][256]   16384 B
    // total 215104 B  (< 320 KB / WGP)

    const int tid  = threadIdx.x;
    const int lane = tid & 31;
    const int wid  = tid >> 5;            // 0..7
    const int qb   = blockIdx.x;          // 0..127
    const int h    = blockIdx.y;
    const int b    = blockIdx.z;
    const size_t headOff = ((size_t)(b * NHEADS + h)) * SEQ * DK;
    const int q0 = qb * 16;

    const int rl = lane & 15;
    const int cq = lane >> 4;

    // ---- load Q tile 16x64 into LDS ----
    {
        int r = tid >> 4;              // 0..15
        int d = (tid & 15) * 4;        // 0..60
        *(uint2*)(Qt + r * DK + d) =
            *(const uint2*)(Qh + headOff + (size_t)(q0 + r) * DK + d);
    }
    __syncthreads();

    // ---- Phase 1: scores = Q K^T / 8 via WMMA; wave w owns keys [w*256,+256) ----
    v16bf aq0, aq1;
    {
        const u16* qp = Qt + rl * DK;
        aq0 = cat16(*(const bf8*)(qp + 8 * cq),      *(const bf8*)(qp + 16 + 8 * cq));
        aq1 = cat16(*(const bf8*)(qp + 32 + 8 * cq), *(const bf8*)(qp + 48 + 8 * cq));
    }
    for (int t = 0; t < 16; ++t) {
        int key = wid * 256 + t * 16;
        // B fragment straight from global K (layout [key][d], d contiguous)
        const u16* kp = Kh + headOff + (size_t)(key + rl) * DK + 8 * cq;
        v16bf bk0 = cat16(*(const bf8*)(kp),      *(const bf8*)(kp + 16));
        v16bf bk1 = cat16(*(const bf8*)(kp + 32), *(const bf8*)(kp + 48));
        v8f a = {0.f,0.f,0.f,0.f,0.f,0.f,0.f,0.f};
        a = wmma_bf16(aq0, bk0, a);     // d = 0..31
        a = wmma_bf16(aq1, bk1, a);     // d = 32..63
        int cc = key + rl;
#pragma unroll
        for (int i = 0; i < 8; ++i)
            sc[(i + 8 * cq) * SEQ + cc] = a[i] * 0.125f;   // 1/sqrt(64)
    }
    __syncthreads();

    // ---- Phase 2: exact k-th largest via 4-pass byte-radix select + exp ----
    // Wave w owns queries {2w, 2w+1}; its histograms are wave-private, and LDS
    // ops of a wave are processed in order, so no extra barriers are needed.
    for (int qq = 0; qq < 2; ++qq) {
        int q = wid * 2 + qq;
        float* row = sc + q * SEQ;
        u32* hq = hist + q * 256;

        // row max (for stable exp); sparse rows contain zeros -> clamp to >= 0
        float m = -1e30f;
        for (int j = lane; j < SEQ; j += 32) m = fmaxf(m, row[j]);
#pragma unroll
        for (int o = 16; o; o >>= 1) m = fmaxf(m, __shfl_xor(m, o));
        m = fmaxf(m, 0.f);

        // byte-radix select of the KTOP-th largest ordered-uint value
        u32 prefix = 0u;
        int want = KTOP;
#pragma unroll
        for (int pass = 0; pass < 4; ++pass) {
            const int sh = 24 - 8 * pass;
            for (int j = lane; j < 256; j += 32) hq[j] = 0u;
            for (int j = lane; j < SEQ; j += 32) {
                u32 u = ordf(row[j]);
                bool match = (((unsigned long long)u >> (sh + 8)) ==
                              ((unsigned long long)prefix >> (sh + 8)));
                if (match) atomicAdd(&hq[(u >> sh) & 255u], 1u);
            }
            u32 bsel = 0u; int w2 = want;
            if (lane == 0) {
                int cum = 0;
                for (int bin = 255; bin >= 0; --bin) {
                    int c = (int)hq[bin];
                    if (cum + c >= want) { bsel = (u32)bin; w2 = want - cum; break; }
                    cum += c;
                }
            }
            bsel = (u32)__shfl((int)bsel, 0);
            want = __shfl(w2, 0);
            prefix |= bsel << sh;
        }
        const u32 val = prefix;   // ordered-uint of the k-th largest score

        // unnormalized probabilities: p = exp((s>=thr ? s : 0) - m)
        float s = 0.f;
        for (int j = lane; j < SEQ; j += 32) {
            float v = row[j];
            float sp = (ordf(v) >= val) ? v : 0.f;
            float p = __expf(sp - m);
            row[j] = p;
            s += p;
        }
#pragma unroll
        for (int o = 16; o; o >>= 1) s += __shfl_xor(s, o);
        if (lane == 0) invden[q] = 1.f / s;
    }
    __syncthreads();

    // ---- Phase 3: out = P @ V via WMMA; wave w accumulates keys [w*256,+256) ----
    v8f acc[4];
    {
        const v8f vz = {0.f,0.f,0.f,0.f,0.f,0.f,0.f,0.f};
        acc[0] = vz; acc[1] = vz; acc[2] = vz; acc[3] = vz;
    }
    u16* vt = Vt + wid * (DK * 32);
    for (int ch = 0; ch < 8; ++ch) {
        int kb = wid * 256 + ch * 32;
        // transpose-stage V chunk (32 keys x 64 dims) into per-wave LDS [d][key]
        const u16* vp = Vh + headOff + (size_t)(kb + lane) * DK;
#pragma unroll
        for (int i = 0; i < 8; ++i) {
            us8 vv = *(const us8*)(vp + 8 * i);
#pragma unroll
            for (int j = 0; j < 8; ++j)
                vt[(8 * i + j) * 32 + lane] = vv[j];
        }
        __syncthreads();   // uniform: every wave does 8 iterations

        // A fragment: probabilities row rl, keys kb + {8cq..,16+8cq..} -> bf16
        const float* pr = sc + rl * SEQ + kb;
        union { u16 u[16]; v16bf v; } cv;
#pragma unroll
        for (int j = 0; j < 8; ++j) cv.u[j]     = f2bf(pr[8 * cq + j]);
#pragma unroll
        for (int j = 0; j < 8; ++j) cv.u[8 + j] = f2bf(pr[16 + 8 * cq + j]);
        v16bf ap = cv.v;

#pragma unroll
        for (int ct = 0; ct < 4; ++ct) {
            const u16* bp = vt + (ct * 16 + rl) * 32 + 8 * cq;
            v16bf bv = cat16(*(const bf8*)(bp), *(const bf8*)(bp + 16));
            acc[ct] = wmma_bf16(ap, bv, acc[ct]);
        }
    }

    // per-wave partial 16x64 -> LDS
    float* pw = partial + wid * (16 * DK);
#pragma unroll
    for (int ct = 0; ct < 4; ++ct)
#pragma unroll
        for (int i = 0; i < 8; ++i)
            pw[(i + 8 * cq) * DK + ct * 16 + rl] = acc[ct][i];
    __syncthreads();

    // cross-wave reduce, normalize, write bf16 ctx [b][s][1024]
    for (int e = tid; e < 16 * DK; e += 256) {
        int r = e >> 6, d = e & 63;
        float sres = 0.f;
#pragma unroll
        for (int w = 0; w < 8; ++w) sres += partial[w * (16 * DK) + e];
        sres *= invden[r];
        ctx[((size_t)(b * SEQ + q0 + r)) * D_MODEL + h * DK + d] = f2bf(sres);
    }
}

// ---------------------------------------------------------------------------
// Host-side launch
// ---------------------------------------------------------------------------
extern "C" void kernel_launch(void* const* d_in, const int* in_sizes, int n_in,
                              void* d_out, int out_size, void* d_ws, size_t ws_size,
                              hipStream_t stream) {
    (void)in_sizes; (void)n_in; (void)out_size; (void)ws_size;
    const float* query  = (const float*)d_in[0];
    const float* key_in = (const float*)d_in[1];
    const float* value  = (const float*)d_in[2];
    const float* W_q    = (const float*)d_in[3];
    const float* b_q    = (const float*)d_in[4];
    const float* W_k    = (const float*)d_in[5];
    const float* b_k    = (const float*)d_in[6];
    const float* W_v    = (const float*)d_in[7];
    const float* b_v    = (const float*)d_in[8];
    const float* W_o    = (const float*)d_in[9];
    const float* b_o    = (const float*)d_in[10];

    char* ws = (char*)d_ws;
    const size_t MB = 1024ull * 1024ull;
    u16* qbf = (u16*)(ws + 0 * MB);     // 16 MB each
    u16* kbf = (u16*)(ws + 16 * MB);
    u16* vbf = (u16*)(ws + 32 * MB);
    u16* wqb = (u16*)(ws + 48 * MB);    // 2 MB each
    u16* wkb = (u16*)(ws + 50 * MB);
    u16* wvb = (u16*)(ws + 52 * MB);
    u16* wob = (u16*)(ws + 54 * MB);
    u16* Qh  = (u16*)(ws + 56 * MB);    // 16 MB each, head-major bf16
    u16* Kh  = (u16*)(ws + 72 * MB);
    u16* Vh  = (u16*)(ws + 88 * MB);
    u16* ctx = (u16*)(ws + 104 * MB);   // 16 MB, [b][s][1024] bf16

    const int nAct = MTOT * D_MODEL;        // 8388608
    const int nW   = D_MODEL * D_MODEL;     // 1048576

    cvt_kernel<<<(nAct + 255) / 256, 256, 0, stream>>>(query,  qbf, nAct);
    cvt_kernel<<<(nAct + 255) / 256, 256, 0, stream>>>(key_in, kbf, nAct);
    cvt_kernel<<<(nAct + 255) / 256, 256, 0, stream>>>(value,  vbf, nAct);
    cvt_kernel<<<(nW   + 255) / 256, 256, 0, stream>>>(W_q, wqb, nW);
    cvt_kernel<<<(nW   + 255) / 256, 256, 0, stream>>>(W_k, wkb, nW);
    cvt_kernel<<<(nW   + 255) / 256, 256, 0, stream>>>(W_v, wvb, nW);
    cvt_kernel<<<(nW   + 255) / 256, 256, 0, stream>>>(W_o, wob, nW);

    dim3 ggrid(D_MODEL / 64, MTOT / 128);   // (16, 64)
    gemm_bf16<<<ggrid, 256, 0, stream>>>(qbf, wqb, b_q, (void*)Qh, 0);
    gemm_bf16<<<ggrid, 256, 0, stream>>>(kbf, wkb, b_k, (void*)Kh, 0);
    gemm_bf16<<<ggrid, 256, 0, stream>>>(vbf, wvb, b_v, (void*)Vh, 0);

    const size_t shmem = 215104;   // see attn_kernel layout
    (void)hipFuncSetAttribute((const void*)attn_kernel,
                              hipFuncAttributeMaxDynamicSharedMemorySize, (int)shmem);
    attn_kernel<<<dim3(SEQ / 16, NHEADS, BATCH), 256, shmem, stream>>>(Qh, Kh, Vh, ctx);

    gemm_bf16<<<ggrid, 256, 0, stream>>>(ctx, wob, b_o, d_out, 1);
}
